// Head_36756330119198
// MI455X (gfx1250) — compile-verified, matrix operand order checked by reading, not run on hardware
//
#include <hip/hip_runtime.h>
#include <hip/hip_bf16.h>
#include <stdint.h>

// ---------------- problem constants ----------------
constexpr int NTOK   = 8192;      // B*S = 4*2048
constexpr int DMODEL = 1024;
constexpr int VOC    = 50257;
constexpr int VPAD   = 50304;     // ceil(VOC/128)*128
constexpr int BM     = 128;
constexpr int BN     = 128;
constexpr int KB     = 32;        // K per WMMA step (bf16 16x16x32)
constexpr int LDS_ROW = 80;       // 32 bf16 (64B) + 16B pad -> conflict-free b128 reads
constexpr int TILE_BYTES = BM * LDS_ROW;   // 10240 B per tile stage

typedef __attribute__((ext_vector_type(16))) __bf16 v16bf;
typedef __attribute__((ext_vector_type(8)))  float  v8f;
typedef __attribute__((ext_vector_type(4)))  unsigned int u32x4;
typedef __attribute__((ext_vector_type(8)))  int    i32x8;
typedef __attribute__((ext_vector_type(4)))  int    i32x4;

union AFrag { v16bf v; uint4 q[2]; };

__device__ __forceinline__ unsigned short f2bf(float f) {
    unsigned u = __float_as_uint(f);
    unsigned r = u + 0x7FFFu + ((u >> 16) & 1u);     // round-to-nearest-even
    return (unsigned short)(r >> 16);
}

// ---------------- TDM: 2D tile load, bf16 elements, padded LDS rows ----------------
// D# layout per CDNA5 ISA §8 (group0 128b, group1 256b). data_size=2B,
// pad_enable=1, pad_interval code 3 (16 DWORDs), pad_amount code 3 (4 DWORDs)
// -> LDS row stride = 64B data + 16B pad = 80B.
__device__ __forceinline__ void tdm_load_2d(uint64_t gaddr, unsigned lds_addr,
                                            unsigned td0, unsigned td1,
                                            unsigned stride_elems,
                                            unsigned tile_d0, unsigned tile_d1) {
    u32x4 g0;
    g0.x = 1u;                                           // count=1, user mode
    g0.y = lds_addr;                                     // LDS byte address
    g0.z = (unsigned)(gaddr & 0xFFFFFFFFull);            // global_addr[31:0]
    g0.w = (unsigned)((gaddr >> 32) & 0x01FFFFFFull) | (2u << 30);  // [56:32] | type=2
    i32x8 g1;
    g1[0] = (int)((1u << 16) | (1u << 20) | (3u << 22) | (3u << 25)); // 2B, pad on
    g1[1] = (int)((td0 & 0xFFFFu) << 16);                             // tensor_dim0 lo16
    g1[2] = (int)(((td0 >> 16) & 0xFFFFu) | ((td1 & 0xFFFFu) << 16)); // d0 hi | d1 lo
    g1[3] = (int)(((td1 >> 16) & 0xFFFFu) | (tile_d0 << 16));         // d1 hi | tile_dim0
    g1[4] = (int)(tile_d1 & 0xFFFFu);                                 // tile_dim1, tile_dim2=0
    g1[5] = (int)stride_elems;                                        // tensor_dim0_stride lo32
    g1[6] = 0;
    g1[7] = 0;
    i32x4 z4; z4[0] = z4[1] = z4[2] = z4[3] = 0;
    i32x8 z8;
    #pragma unroll
    for (int i = 0; i < 8; ++i) z8[i] = 0;
    // amdgpu-toolchain (clang-23 / therock-10.0) 6-arg form
    __builtin_amdgcn_tensor_load_to_lds(g0, g1, z4, z4, z8, 0);
}

// ---------------- kernel 0: zero loss accumulators ----------------
__global__ void k_init(float* accum) { accum[0] = 0.f; accum[1] = 0.f; }

// ---------------- kernel 1: LayerNorm -> bf16 ----------------
__global__ __launch_bounds__(256) void k_layernorm(const float* __restrict__ x,
                                                   const float* __restrict__ gamma,
                                                   const float* __restrict__ beta,
                                                   unsigned short* __restrict__ hbf) {
    const int r = blockIdx.x;
    const int t = threadIdx.x;
    const float4 v = ((const float4*)(x + (size_t)r * DMODEL))[t];
    float sum = v.x + v.y + v.z + v.w;
    float sq  = v.x*v.x + v.y*v.y + v.z*v.z + v.w*v.w;
    #pragma unroll
    for (int off = 16; off; off >>= 1) {
        sum += __shfl_xor(sum, off, 32);
        sq  += __shfl_xor(sq,  off, 32);
    }
    __shared__ float s1[8], s2[8], bc[2];
    const int w = t >> 5, l = t & 31;
    if (l == 0) { s1[w] = sum; s2[w] = sq; }
    __syncthreads();
    if (t == 0) {
        float a = 0.f, b = 0.f;
        #pragma unroll
        for (int i = 0; i < 8; ++i) { a += s1[i]; b += s2[i]; }
        bc[0] = a * (1.f / DMODEL);
        bc[1] = b * (1.f / DMODEL);
    }
    __syncthreads();
    const float mu  = bc[0];
    const float inv = rsqrtf(bc[1] - mu * mu + 1e-5f);
    const float4 g  = ((const float4*)gamma)[t];
    const float4 bb = ((const float4*)beta)[t];
    const float o0 = (v.x - mu) * inv * g.x + bb.x;
    const float o1 = (v.y - mu) * inv * g.y + bb.y;
    const float o2 = (v.z - mu) * inv * g.z + bb.z;
    const float o3 = (v.w - mu) * inv * g.w + bb.w;
    uint2 out;
    out.x = (unsigned)f2bf(o0) | ((unsigned)f2bf(o1) << 16);
    out.y = (unsigned)f2bf(o2) | ((unsigned)f2bf(o3) << 16);
    ((uint2*)hbf)[(size_t)r * (DMODEL / 4) + t] = out;
}

// ---------------- kernel 2: W f32 -> bf16 (zero-pad rows VOC..VPAD) ----------------
__global__ __launch_bounds__(256) void k_convert_w(const float* __restrict__ W,
                                                   unsigned short* __restrict__ wbf) {
    const long long i = ((long long)blockIdx.x * 256 + threadIdx.x) * 8;
    constexpr long long nreal = (long long)VOC * DMODEL;   // multiple of 8
    uint4 out;
    if (i < nreal) {
        const float4* p = (const float4*)(W + i);
        const float4 v0 = p[0], v1 = p[1];
        out.x = (unsigned)f2bf(v0.x) | ((unsigned)f2bf(v0.y) << 16);
        out.y = (unsigned)f2bf(v0.z) | ((unsigned)f2bf(v0.w) << 16);
        out.z = (unsigned)f2bf(v1.x) | ((unsigned)f2bf(v1.y) << 16);
        out.w = (unsigned)f2bf(v1.z) | ((unsigned)f2bf(v1.w) << 16);
    } else {
        out = make_uint4(0, 0, 0, 0);
    }
    *(uint4*)(wbf + i) = out;
}

// ---------------- kernel 3: bf16 WMMA GEMM, TDM double-buffered tiles ----------------
__global__ __launch_bounds__(256) void k_gemm_wmma(const unsigned short* __restrict__ hbf,
                                                   const unsigned short* __restrict__ wbf,
                                                   float* __restrict__ logits) {
    __shared__ __align__(16) unsigned char ldsA[2][TILE_BYTES];
    __shared__ __align__(16) unsigned char ldsB[2][TILE_BYTES];

    const int tid  = threadIdx.x;
    const int wave = tid >> 5;
    const int lane = tid & 31;
    const int m0 = blockIdx.y * BM;
    const int n0 = blockIdx.x * BN;
    const int wm = (wave >> 2) * 64;   // 2 waves along M
    const int wn = (wave & 3) * 32;    // 4 waves along N
    const int lrow = lane & 15;
    const int kg   = (lane >> 4) * 16; // byte offset: K-elems 0..7 vs 8..15

    v8f acc[4][2];
    #pragma unroll
    for (int i = 0; i < 4; ++i)
        #pragma unroll
        for (int j = 0; j < 2; ++j)
            acc[i][j] = (v8f){0.f, 0.f, 0.f, 0.f, 0.f, 0.f, 0.f, 0.f};

    const uint64_t abase = (uint64_t)(uintptr_t)hbf;
    const uint64_t bbase = (uint64_t)(uintptr_t)wbf;

    auto issue = [&](int kt, int st) {
        const uint64_t ga = abase + (((uint64_t)m0 * DMODEL + (uint64_t)kt * KB) << 1);
        const uint64_t gb = bbase + (((uint64_t)n0 * DMODEL + (uint64_t)kt * KB) << 1);
        tdm_load_2d(ga, (unsigned)(uintptr_t)&ldsA[st][0], DMODEL, NTOK, DMODEL, KB, BM);
        tdm_load_2d(gb, (unsigned)(uintptr_t)&ldsB[st][0], DMODEL, VPAD, DMODEL, KB, BN);
    };

    constexpr int NK = DMODEL / KB;    // 32
    if (wave == 0) issue(0, 0);

    for (int kt = 0; kt < NK; ++kt) {
        const int st = kt & 1;
        if (wave == 0) {
            if (kt + 1 < NK) { issue(kt + 1, st ^ 1); __builtin_amdgcn_s_wait_tensorcnt(2); }
            else             {                         __builtin_amdgcn_s_wait_tensorcnt(0); }
        }
        __syncthreads();

        AFrag a[4], b[2];
        #pragma unroll
        for (int mi = 0; mi < 4; ++mi) {
            const unsigned char* p = &ldsA[st][(wm + mi * 16 + lrow) * LDS_ROW + kg];
            a[mi].q[0] = *(const uint4*)p;
            a[mi].q[1] = *(const uint4*)(p + 32);
        }
        #pragma unroll
        for (int ni = 0; ni < 2; ++ni) {
            const unsigned char* p = &ldsB[st][(wn + ni * 16 + lrow) * LDS_ROW + kg];
            b[ni].q[0] = *(const uint4*)p;
            b[ni].q[1] = *(const uint4*)(p + 32);
        }
        #pragma unroll
        for (int mi = 0; mi < 4; ++mi)
            #pragma unroll
            for (int ni = 0; ni < 2; ++ni)
                acc[mi][ni] = __builtin_amdgcn_wmma_f32_16x16x32_bf16(
                    false, a[mi].v, false, b[ni].v, (short)0, acc[mi][ni], false, false);
        __syncthreads();
    }

    // C/D layout: lanes 0-15 -> M=r, N=lane; lanes 16-31 -> M=r+8, N=lane-16
    const int mlo = (lane >> 4) * 8;
    #pragma unroll
    for (int mi = 0; mi < 4; ++mi) {
        #pragma unroll
        for (int ni = 0; ni < 2; ++ni) {
            const int col = n0 + wn + ni * 16 + lrow;
            if (col < VOC) {
                const size_t base = (size_t)(m0 + wm + mi * 16 + mlo) * VOC + col;
                #pragma unroll
                for (int r = 0; r < 8; ++r)
                    logits[base + (size_t)r * VOC] = acc[mi][ni][r];
            }
        }
    }
}

// ---------------- kernel 4: online softmax + masked NLL ----------------
__global__ __launch_bounds__(256) void k_softmax_loss(const float* __restrict__ logits,
                                                      const int* __restrict__ targets,
                                                      float* __restrict__ accum) {
    const int r = blockIdx.x;
    const float* row = logits + (size_t)r * VOC;
    float m = -3.4e38f, s = 0.f;
    for (int j = threadIdx.x; j < VOC; j += 256) {
        const float x = row[j];
        const float nm = fmaxf(m, x);
        s = s * __expf(m - nm) + __expf(x - nm);
        m = nm;
    }
    #pragma unroll
    for (int off = 16; off; off >>= 1) {
        const float mo = __shfl_xor(m, off, 32);
        const float so = __shfl_xor(s, off, 32);
        const float nm = fmaxf(m, mo);
        s = s * __expf(m - nm) + so * __expf(mo - nm);
        m = nm;
    }
    __shared__ float sm[8], ss[8];
    const int w = threadIdx.x >> 5, l = threadIdx.x & 31;
    if (l == 0) { sm[w] = m; ss[w] = s; }
    __syncthreads();
    if (threadIdx.x == 0) {
        m = sm[0]; s = ss[0];
        #pragma unroll
        for (int i = 1; i < 8; ++i) {
            const float nm = fmaxf(m, sm[i]);
            s = s * __expf(m - nm) + ss[i] * __expf(sm[i] - nm);
            m = nm;
        }
        const int t = targets[r];
        if (t != 0) {
            const float nll = (m + __logf(s)) - row[t];
            atomicAdd(&accum[0], nll);
            atomicAdd(&accum[1], 1.0f);
        }
    }
}

__global__ void k_finalize(const float* accum, float* loss_out) {
    loss_out[0] = accum[0] / fmaxf(accum[1], 1.0f);
}

// ---------------- launch ----------------
extern "C" void kernel_launch(void* const* d_in, const int* in_sizes, int n_in,
                              void* d_out, int out_size, void* d_ws, size_t ws_size,
                              hipStream_t stream) {
    const float* context = (const float*)d_in[0];
    const int*   targets = (const int*)d_in[1];
    const float* gamma   = (const float*)d_in[2];
    const float* beta    = (const float*)d_in[3];
    const float* W       = (const float*)d_in[4];

    float* logits = (float*)d_out;
    float* lossp  = logits + (size_t)NTOK * VOC;

    // ws layout: [0..255] loss accumulators | h_bf16 (16 MB) | W_bf16 padded (~103 MB)
    float* accum = (float*)d_ws;
    unsigned short* hbf = (unsigned short*)((char*)d_ws + 256);
    unsigned short* wbf = hbf + (size_t)NTOK * DMODEL;

    k_init<<<1, 1, 0, stream>>>(accum);
    k_layernorm<<<NTOK, 256, 0, stream>>>(context, gamma, beta, hbf);
    k_convert_w<<<((long long)VPAD * DMODEL) / 2048, 256, 0, stream>>>(W, wbf);
    dim3 grid(VPAD / BN, NTOK / BM);   // 393 x 64
    k_gemm_wmma<<<grid, 256, 0, stream>>>(hbf, wbf, logits);
    k_softmax_loss<<<NTOK, 256, 0, stream>>>(logits, targets, accum);
    k_finalize<<<1, 1, 0, stream>>>(accum, lossp);
}
